// NextVLAD_33913061769335
// MI455X (gfx1250) — matrix operand order repaired and checked by reading, not compiled, not run on hardware
//
#include <hip/hip_runtime.h>
#include <hip/hip_bf16.h>
#include <math.h>

typedef __attribute__((ext_vector_type(16))) __bf16 v16bf;
typedef __attribute__((ext_vector_type(8)))  __bf16 v8bf;
typedef __attribute__((ext_vector_type(8)))  float  v8f;
typedef __attribute__((ext_vector_type(2)))  float  v2f;

#define NBATCH 8
#define HW     196
#define MTOK   1568      // 8*196 tokens per batch
#define TOK    12544     // 64*196 tokens total
#define CIN    768
#define N2     1536
#define GRP    6
#define KC     128
#define DD     256
#define NOUT   48
#define NTRI   1176

// ---------------------------------------------------------------- fragment IO
// 16-bit A-matrix 16x32: lane half h holds contiguous K in [h*16, h*16+16)
// (matches B fragment order).
static __device__ inline v16bf load_afrag(const __bf16* rowp, int kk, int half) {
  return *(const v16bf*)(rowp + kk + half * 16);
}
// B-matrix 32x16: lane = column n, contiguous K chunk of 16.
static __device__ inline v16bf load_bfrag(const __bf16* colp, int kk, int half) {
  return *(const v16bf*)(colp + kk + half * 16);
}

// ---------------------------------------------------------------- 1) L2 norms
__global__ void k_norm(const float* __restrict__ x, float* __restrict__ rnorm) {
  int b8 = blockIdx.x, hw = threadIdx.x;
  if (hw >= HW) return;
  const float* p = x + (size_t)b8 * CIN * HW + hw;
  float ss = 0.f;
  for (int c = 0; c < CIN; ++c) { float v = p[(size_t)c * HW]; ss += v * v; }
  rnorm[b8 * HW + hw] = 1.0f / fmaxf(sqrtf(ss), 1e-12f);
}

// ------------------------------------------- 2) transpose+normalize+cvt bf16
// token row = b8*196 + hw (the clip-of-8 regroup collapses to exactly this)
__global__ void k_pack_x(const float* __restrict__ x, const float* __restrict__ rnorm,
                         __bf16* __restrict__ xn) {
  __shared__ float tile[16][17];
  int b8 = blockIdx.x, cb = blockIdx.y, hb = blockIdx.z;
  int tx = threadIdx.x, ty = threadIdx.y;
  int hw = hb * 16 + tx;
  if (hw < HW) tile[ty][tx] = x[((size_t)b8 * CIN + cb * 16 + ty) * HW + hw];
  __syncthreads();
  int hw2 = hb * 16 + ty;
  if (hw2 < HW) {
    int row = b8 * HW + hw2;
    xn[(size_t)row * CIN + cb * 16 + tx] = (__bf16)(tile[tx][ty] * rnorm[row]);
  }
}

// ---------------------------------------------------------------- 3) f32->bf16
__global__ void k_cvt_bf16(const float* __restrict__ in, __bf16* __restrict__ out, int n) {
  int i = blockIdx.x * 256 + threadIdx.x;
  if (i < n) out[i] = (__bf16)in[i];
}

// -------------------------------------- 4) generic bf16 GEMM  C = A*B^T + bias
// A[M][K] bf16, B[N][K] bf16 (K-contiguous), block = 64x64 tile, 8 waves.
__global__ __launch_bounds__(256)
void k_gemm_bf16(const __bf16* __restrict__ A, const __bf16* __restrict__ B,
                 const float* __restrict__ bias, float* __restrict__ Cf,
                 __bf16* __restrict__ Cb, int N, int K) {
  int tid = threadIdx.x, lane = tid & 31, wave = tid >> 5;
  int wr = wave & 3, wc = wave >> 2;
  int half = lane >> 4, lr = lane & 15;
  int m0 = blockIdx.x * 64 + wr * 16;
  int n0 = blockIdx.y * 64 + wc * 32;
  const __bf16* Ap  = A + (size_t)(m0 + lr) * K;
  const __bf16* Bp0 = B + (size_t)(n0 + lr) * K;
  const __bf16* Bp1 = Bp0 + (size_t)16 * K;
  v8f c0 = {}, c1 = {};
  for (int kk = 0; kk < K; kk += 32) {
    v16bf a  = load_afrag(Ap,  kk, half);
    v16bf b0 = load_bfrag(Bp0, kk, half);
    v16bf b1 = load_bfrag(Bp1, kk, half);
    c0 = __builtin_amdgcn_wmma_f32_16x16x32_bf16(false, a, false, b0, (short)0, c0, false, false);
    c1 = __builtin_amdgcn_wmma_f32_16x16x32_bf16(false, a, false, b1, (short)0, c1, false, false);
  }
  int mrow = m0 + half * 8;
  int nc0 = n0 + lr, nc1 = n0 + 16 + lr;
  float bi0 = bias ? bias[nc0] : 0.f, bi1 = bias ? bias[nc1] : 0.f;
  for (int r = 0; r < 8; ++r) {
    size_t i0 = (size_t)(mrow + r) * N + nc0;
    size_t i1 = (size_t)(mrow + r) * N + nc1;
    float v0 = c0[r] + bi0, v1 = c1[r] + bi1;
    Cf[i0] = v0; Cf[i1] = v1;
    if (Cb) { Cb[i0] = (__bf16)v0; Cb[i1] = (__bf16)v1; }
  }
}

// ------------------------------------------------- 5) bf16 transpose (for Y^T)
__global__ void k_transpose_bf16(const __bf16* __restrict__ in, __bf16* __restrict__ out,
                                 int M, int N) {  // in [M][N] -> out [N][M]
  __shared__ __bf16 tile[32][33];
  int bx = blockIdx.x, by = blockIdx.y;
  int tx = threadIdx.x, ty = threadIdx.y;
  for (int i = ty; i < 32; i += 8)
    tile[i][tx] = in[(size_t)(bx * 32 + i) * N + by * 32 + tx];
  __syncthreads();
  for (int i = ty; i < 32; i += 8)
    out[(size_t)(by * 32 + i) * M + bx * 32 + tx] = tile[tx][i];
}

// ---------------------------------------------------------------- 6) alpha_g
__global__ void k_alpha_g(const float* __restrict__ Y, const float* __restrict__ Wg,
                          const float* __restrict__ bg, float* __restrict__ ag) {
  __shared__ float red[256];
  int row = blockIdx.x, tid = threadIdx.x;
  float acc[GRP]; for (int g = 0; g < GRP; ++g) acc[g] = 0.f;
  for (int j = tid; j < N2; j += 256) {
    float yv = Y[(size_t)row * N2 + j];
    for (int g = 0; g < GRP; ++g) acc[g] += yv * Wg[g * N2 + j];
  }
  for (int g = 0; g < GRP; ++g) {
    red[tid] = acc[g]; __syncthreads();
    for (int s = 128; s > 0; s >>= 1) { if (tid < s) red[tid] += red[tid + s]; __syncthreads(); }
    if (tid == 0) ag[row * GRP + g] = 1.0f / (1.0f + __expf(-(red[0] + bg[g])));
    __syncthreads();
  }
}

// -------------------- 7) softmax over tokens (dim=1), fused *alpha_g, w^T bf16
__global__ void k_softmax(const float* __restrict__ L, const float* __restrict__ ag,
                          __bf16* __restrict__ wT, float* __restrict__ wsum) {
  __shared__ float red[8][33];
  __shared__ float smax[32], ssum[32];
  int jb = blockIdx.x, b = blockIdx.y;
  int tx = threadIdx.x, ty = threadIdx.y;
  int j = jb * 32 + tx, g = j >> 7, kidx = j & 127;
  float mx = -1e30f;
  for (int m = ty; m < MTOK; m += 8)
    mx = fmaxf(mx, L[(size_t)(b * MTOK + m) * (GRP * KC) + j]);
  red[ty][tx] = mx; __syncthreads();
  if (ty == 0) { float v = red[0][tx]; for (int r = 1; r < 8; ++r) v = fmaxf(v, red[r][tx]); smax[tx] = v; }
  __syncthreads();
  float mxv = smax[tx], s = 0.f;
  for (int m = ty; m < MTOK; m += 8)
    s += __expf(L[(size_t)(b * MTOK + m) * (GRP * KC) + j] - mxv);
  red[ty][tx] = s; __syncthreads();
  if (ty == 0) { float v = 0.f; for (int r = 0; r < 8; ++r) v += red[r][tx]; ssum[tx] = v; }
  __syncthreads();
  float inv = 1.0f / ssum[tx], ws = 0.f;
  for (int m = ty; m < MTOK; m += 8) {
    int row = b * MTOK + m;
    float w = __expf(L[(size_t)row * (GRP * KC) + j] - mxv) * inv * ag[row * GRP + g];
    wT[((size_t)((b * GRP + g) * KC + kidx)) * MTOK + m] = (__bf16)w;
    ws += w;
  }
  red[ty][tx] = ws; __syncthreads();
  if (ty == 0) { float v = 0.f; for (int r = 0; r < 8; ++r) v += red[r][tx]; wsum[b * (GRP * KC) + j] = v; }
}

// -------------- 8) VLAD: vlad[b,g,k,d] = sum_m wT[k,m]*YT[d,m] - wsum*centroid
__global__ __launch_bounds__(256)
void k_vlad(const __bf16* __restrict__ wT, const __bf16* __restrict__ YT,
            const float* __restrict__ wsum, const float* __restrict__ cent,
            float* __restrict__ vlad) {
  int bid = blockIdx.x;           // 96 blocks: b(8) x g(6) x dblk(2)
  int b = bid / 12, g = (bid % 12) / 2, db = bid & 1;
  int tid = threadIdx.x, lane = tid & 31, wave = tid >> 5;
  int half = lane >> 4, lr = lane & 15;
  int krow = wave * 16 + lr;
  const __bf16* Ap = wT + ((size_t)((b * GRP + g) * KC + krow)) * MTOK;
  const v8f vz = {};
  v8f c[8]; for (int j = 0; j < 8; ++j) c[j] = vz;
  for (int mm = 0; mm < MTOK; mm += 32) {
    v16bf a = load_afrag(Ap, mm, half);
    for (int j = 0; j < 8; ++j) {
      int d = db * 128 + j * 16 + lr;
      v16bf bb = *(const v16bf*)(YT + ((size_t)(g * DD + d)) * TOK + b * MTOK + mm + half * 16);
      c[j] = __builtin_amdgcn_wmma_f32_16x16x32_bf16(false, a, false, bb, (short)0, c[j], false, false);
    }
  }
  int kbase = wave * 16 + half * 8;
  for (int j = 0; j < 8; ++j)
    for (int r = 0; r < 8; ++r) {
      int k = kbase + r, d = db * 128 + j * 16 + lr;
      float v = c[j][r] - wsum[b * (GRP * KC) + g * KC + k] * cent[k * DD + d];
      vlad[((size_t)((b * GRP + g) * KC + k)) * DD + d] = v;
    }
}

// -------------------------- 9) vlad @ W_red^T + b_red -> r[b][k][o][g] (vk)
__global__ void k_reduce(const float* __restrict__ vlad, const float* __restrict__ Wr,
                         const float* __restrict__ br, float* __restrict__ rr) {
  __shared__ float sv[DD];
  int bid = blockIdx.x;           // b*6*128
  int b = bid / (GRP * KC), g = (bid / KC) % GRP, k = bid % KC;
  int tid = threadIdx.x;
  sv[tid] = vlad[((size_t)((b * GRP + g) * KC + k)) * DD + tid];
  __syncthreads();
  if (tid < NOUT) {
    float acc = br[tid];
    for (int d = 0; d < DD; ++d) acc += sv[d] * Wr[tid * DD + d];
    rr[((size_t)((b * KC + k) * NOUT + tid)) * GRP + g] = acc;
  }
}

// ------------------------- 10) covpool + Newton-Schulz(3) via f32 WMMA 16x16x4
static __device__ inline void mm48(const float* As, const float* Bs, v8f* acc, int lane) {
  const int half = lane >> 4, lr = lane & 15;
  const v8f vz = {};
  for (int f = 0; f < 9; ++f) acc[f] = vz;
  for (int kk = 0; kk < 48; kk += 4) {
    int k0 = kk + half * 2;
    v2f a[3], bv[3];
    for (int i = 0; i < 3; ++i) {
      const float* ar = As + (i * 16 + lr) * 48 + k0;
      a[i].x = ar[0]; a[i].y = ar[1];
      bv[i].x = Bs[k0 * 48 + i * 16 + lr];
      bv[i].y = Bs[(k0 + 1) * 48 + i * 16 + lr];
    }
    for (int i = 0; i < 3; ++i)
      for (int j = 0; j < 3; ++j)
        acc[i * 3 + j] = __builtin_amdgcn_wmma_f32_16x16x4_f32(
            false, a[i], false, bv[j], (short)0, acc[i * 3 + j], false, false);
  }
}
// store: D[r][c] = mul*acc + (r==c ? dadd : 0)
static __device__ inline void st48(float* D, const v8f* acc, int lane, float mul, float dadd) {
  const int half = lane >> 4, lr = lane & 15;
  for (int i = 0; i < 3; ++i)
    for (int j = 0; j < 3; ++j)
      for (int r = 0; r < 8; ++r) {
        int row = i * 16 + half * 8 + r, col = j * 16 + lr;
        float v = acc[i * 3 + j][r] * mul;
        if (row == col) v += dadd;
        D[row * 48 + col] = v;
      }
}

__global__ __launch_bounds__(64)
void k_cov_ns(const float* __restrict__ rr, float* __restrict__ out) {
  __shared__ float bufs[2][3][48 * 48];
  int tid = threadIdx.x, wv = tid >> 5, lane = tid & 31;
  int mat = blockIdx.x * 2 + wv;               // (b*128 + k), 1024 total
  float* B0 = bufs[wv][0];
  float* B1 = bufs[wv][1];
  float* B2 = bufs[wv][2];
  // load v (48x6) into B2 head
  const float* vp = rr + (size_t)mat * (NOUT * GRP);
  for (int i = lane; i < NOUT * GRP; i += 32) B2[i] = vp[i];
  __syncthreads();
  // covpool: cov = (1/6) v v^T - (1/36) s s^T
  for (int idx = lane; idx < 2304; idx += 32) {
    int i = idx / 48, j = idx % 48;
    float dp = 0.f, si = 0.f, sj = 0.f;
    for (int g = 0; g < GRP; ++g) {
      float a = B2[i * GRP + g], b = B2[j * GRP + g];
      dp += a * b; si += a; sj += b;
    }
    B0[idx] = dp * (1.0f / 6.0f) - si * sj * (1.0f / 36.0f);
  }
  __syncthreads();
  float tr = 0.f;
  for (int i = lane; i < 48; i += 32) tr += B0[i * 48 + i];
  for (int off = 16; off > 0; off >>= 1) tr += __shfl_xor(tr, off);
  float itr = 1.0f / tr;
  // An -> B0, T1 = 0.5*(3I - An) -> B1
  for (int idx = lane; idx < 2304; idx += 32) {
    float an = B0[idx] * itr;
    B0[idx] = an;
    int i = idx / 48, j = idx % 48;
    B1[idx] = 0.5f * ((i == j ? 3.0f : 0.0f) - an);
  }
  __syncthreads();
  v8f acc[9];
  mm48(B0, B1, acc, lane);  __syncthreads(); st48(B2, acc, lane, 1.0f, 0.0f);   __syncthreads(); // Y0
  mm48(B1, B2, acc, lane);  __syncthreads(); st48(B0, acc, lane, -0.5f, 1.5f);  __syncthreads(); // ZY1
  mm48(B2, B0, acc, lane);  __syncthreads(); st48(B2, acc, lane, 1.0f, 0.0f);   __syncthreads(); // Y1
  mm48(B0, B1, acc, lane);  __syncthreads(); st48(B1, acc, lane, 1.0f, 0.0f);   __syncthreads(); // Z1
  mm48(B1, B2, acc, lane);  __syncthreads(); st48(B0, acc, lane, -1.0f, 3.0f);  __syncthreads(); // T2
  mm48(B2, B0, acc, lane);  __syncthreads();
  st48(B1, acc, lane, 0.5f * sqrtf(tr), 0.0f);                                  __syncthreads(); // Yf
  // upper-triangle extraction (row-major triu order)
  float* op = out + (size_t)mat * NTRI;
  for (int t = lane; t < NTRI; t += 32) {
    int i = (int)((97.0f - sqrtf(9409.0f - 8.0f * (float)t)) * 0.5f);
    if (i > 47) i = 47;
    while (i > 0 && (i * (97 - i)) / 2 > t) --i;
    while (((i + 1) * (96 - i)) / 2 <= t) ++i;
    int j = i + (t - (i * (97 - i)) / 2);
    op[t] = B1[i * 48 + j];
  }
}

// ================================================================= launcher
extern "C" void kernel_launch(void* const* d_in, const int* in_sizes, int n_in,
                              void* d_out, int out_size, void* d_ws, size_t ws_size,
                              hipStream_t stream) {
  const float* x      = (const float*)d_in[0];
  const float* cent   = (const float*)d_in[1];
  const float* W_inp  = (const float*)d_in[2];
  const float* b_inp  = (const float*)d_in[3];
  const float* W_g    = (const float*)d_in[4];
  const float* b_g    = (const float*)d_in[5];
  const float* W_gk   = (const float*)d_in[6];
  const float* b_gk   = (const float*)d_in[7];
  const float* W_red  = (const float*)d_in[8];
  const float* b_red  = (const float*)d_in[9];
  float* out = (float*)d_out;

  char* ws = (char*)d_ws;
  size_t off = 0;
  auto alloc = [&](size_t nb) -> void* {
    void* p = ws + off;
    off = (off + nb + 255) & ~(size_t)255;
    return p;
  };
  float*  rnorm = (float*) alloc((size_t)TOK * 4);
  __bf16* xn    = (__bf16*)alloc((size_t)TOK * CIN * 2);
  __bf16* wib   = (__bf16*)alloc((size_t)N2 * CIN * 2);
  __bf16* wgb   = (__bf16*)alloc((size_t)(GRP * KC) * N2 * 2);
  float*  Y     = (float*) alloc((size_t)TOK * N2 * 4);
  __bf16* Ybf   = (__bf16*)alloc((size_t)TOK * N2 * 2);
  __bf16* YT    = (__bf16*)alloc((size_t)N2 * TOK * 2);
  float*  ag    = (float*) alloc((size_t)TOK * GRP * 4);
  float*  Lg    = (float*) alloc((size_t)TOK * (GRP * KC) * 4);
  __bf16* wT    = (__bf16*)alloc((size_t)NBATCH * GRP * KC * MTOK * 2);
  float*  wsum  = (float*) alloc((size_t)NBATCH * GRP * KC * 4);
  float*  vlad  = (float*) alloc((size_t)NBATCH * GRP * KC * DD * 4);
  float*  rr    = (float*) alloc((size_t)NBATCH * KC * NOUT * GRP * 4);

  k_norm<<<64, 256, 0, stream>>>(x, rnorm);
  k_pack_x<<<dim3(64, CIN / 16, 13), dim3(16, 16), 0, stream>>>(x, rnorm, xn);
  k_cvt_bf16<<<(N2 * CIN + 255) / 256, 256, 0, stream>>>(W_inp, wib, N2 * CIN);
  k_cvt_bf16<<<(GRP * KC * N2 + 255) / 256, 256, 0, stream>>>(W_gk, wgb, GRP * KC * N2);
  // GEMM1: Y = xn @ W_inp^T + b_inp   [12544 x 1536], K=768
  k_gemm_bf16<<<dim3(TOK / 64, N2 / 64), 256, 0, stream>>>(xn, wib, b_inp, Y, Ybf, N2, CIN);
  k_transpose_bf16<<<dim3(TOK / 32, N2 / 32), dim3(32, 8), 0, stream>>>(Ybf, YT, TOK, N2);
  k_alpha_g<<<TOK, 256, 0, stream>>>(Y, W_g, b_g, ag);
  // GEMM2: Lg = Y @ W_gk^T + b_gk    [12544 x 768], K=1536
  k_gemm_bf16<<<dim3(TOK / 64, (GRP * KC) / 64), 256, 0, stream>>>(Ybf, wgb, b_gk, Lg, (/*no bf16*/ __bf16*)nullptr, GRP * KC, N2);
  k_softmax<<<dim3((GRP * KC) / 32, NBATCH), dim3(32, 8), 0, stream>>>(Lg, ag, wT, wsum);
  k_vlad<<<96, 256, 0, stream>>>(wT, YT, wsum, cent, vlad);
  k_reduce<<<NBATCH * GRP * KC, 256, 0, stream>>>(vlad, W_red, b_red, rr);
  k_cov_ns<<<512, 64, 0, stream>>>(rr, out);
}
